// HyperMoELayer_38001870635035
// MI455X (gfx1250) — matrix operand order repaired and checked by reading, not compiled, not run on hardware
//
#include <hip/hip_runtime.h>
#include <stdint.h>

// ---- problem constants (match reference) ----
#define D_MODEL 1024
#define D_FF    4096
#define N_EXP   8
#define T_TOK   2048            // B*S
#define CAP     2048            // max tokens routed to one expert (top-k distinct)

#define FF_CHUNK 128            // d_ff columns processed per outer iteration
#define KSTEP    32             // WMMA K per step (bf16 16x16x32)
#define XSTRIDE  (D_MODEL + 8)  // bf16 elems; *2 bytes multiple of 16 -> aligned b128
#define WSTRIDE  (KSTEP + 8)    // 40 bf16 = 80B rows, 16B aligned, conflict-spread
#define HSTRIDE  (FF_CHUNK + 8) // 136 bf16 = 272B rows, 16B aligned

typedef __attribute__((ext_vector_type(16))) __bf16 v16bf;
typedef __attribute__((ext_vector_type(8)))  float  v8f;

union V16BF {
    v16bf v;
    uint4 q[2];
    uint32_t u[8];
};

// fp32 -> bf16 (round-to-nearest-even), packed pair into one dword
__device__ __forceinline__ uint32_t pk_bf16(float a, float b) {
    uint32_t ua = __float_as_uint(a);
    uint32_t ub = __float_as_uint(b);
    ua = (ua + 0x7FFFu + ((ua >> 16) & 1u)) >> 16;
    ub = (ub + 0x7FFFu + ((ub >> 16) & 1u)) >> 16;
    return (ua & 0xFFFFu) | (ub << 16);
}
__device__ __forceinline__ uint16_t f2bf(float a) {
    uint32_t ua = __float_as_uint(a);
    return (uint16_t)((ua + 0x7FFFu + ((ua >> 16) & 1u)) >> 16);
}

// generic pointer to LDS: addr[31:0] IS the LDS byte offset (aperture layout)
__device__ __forceinline__ uint32_t lds_off(const void* p) {
    return (uint32_t)(uintptr_t)p;
}

// CDNA5 async copy: global -> LDS, 16B per lane, tracked by ASYNCcnt
__device__ __forceinline__ void async_ld_b128(uint32_t lds_addr, const void* gaddr) {
    asm volatile("global_load_async_to_lds_b128 %0, %1, off"
                 :: "v"(lds_addr), "v"(gaddr) : "memory");
}
__device__ __forceinline__ void wait_async0() {
#if __has_builtin(__builtin_amdgcn_s_wait_asynccnt)
    __builtin_amdgcn_s_wait_asynccnt(0);
#else
    asm volatile("s_wait_asynccnt 0x0" ::: "memory");
#endif
}

// -------------------------------------------------------------------------
// Kernel 0: bulk fp32 -> bf16 conversion (one pass per launch; amortizes the
// cast across all expert tiles and halves GEMM-side HBM traffic).
// -------------------------------------------------------------------------
__global__ __launch_bounds__(256) void cvt_f32_bf16(
    const float4* __restrict__ src, uint2* __restrict__ dst, int n4)
{
    int i = blockIdx.x * blockDim.x + threadIdx.x;
    int stride = gridDim.x * blockDim.x;
    for (; i < n4; i += stride) {
        float4 f = src[i];
        dst[i] = make_uint2(pk_bf16(f.x, f.y), pk_bf16(f.z, f.w));
    }
}

// -------------------------------------------------------------------------
// Kernel 1: router. One wave32 per token: logits = x . Wr^T (8 experts),
// top-2 + softmax, atomic-append (token, weight) to each chosen expert list.
// -------------------------------------------------------------------------
__global__ __launch_bounds__(256) void moe_router(
    const float* __restrict__ x, const float* __restrict__ wr,
    int* __restrict__ cnt, int* __restrict__ tok, float* __restrict__ wt)
{
    const int wave = threadIdx.x >> 5;
    const int lane = threadIdx.x & 31;
    const int t = blockIdx.x * 8 + wave;
    if (t >= T_TOK) return;

    const float* xr = x + (size_t)t * D_MODEL;
    float acc[N_EXP];
#pragma unroll
    for (int e = 0; e < N_EXP; ++e) acc[e] = 0.f;

    for (int d = lane; d < D_MODEL; d += 32) {
        float xv = xr[d];
#pragma unroll
        for (int e = 0; e < N_EXP; ++e) acc[e] += xv * wr[e * D_MODEL + d];
    }
#pragma unroll
    for (int e = 0; e < N_EXP; ++e) {
#pragma unroll
        for (int off = 16; off > 0; off >>= 1)
            acc[e] += __shfl_xor(acc[e], off, 32);
    }

    if (lane == 0) {
        int i1 = 0; float l1 = acc[0];
#pragma unroll
        for (int e = 1; e < N_EXP; ++e) if (acc[e] > l1) { l1 = acc[e]; i1 = e; }
        int i2 = -1; float l2 = -3.4e38f;
#pragma unroll
        for (int e = 0; e < N_EXP; ++e) if (e != i1 && acc[e] > l2) { l2 = acc[e]; i2 = e; }
        float p1 = 1.f / (1.f + __expf(l2 - l1));
        float p2 = 1.f - p1;
        int s1 = atomicAdd(&cnt[i1], 1);
        tok[i1 * CAP + s1] = t;  wt[i1 * CAP + s1] = p1;
        int s2 = atomicAdd(&cnt[i2], 1);
        tok[i2 * CAP + s2] = t;  wt[i2 * CAP + s2] = p2;
    }
}

// -------------------------------------------------------------------------
// Kernel 2 (fast path): fused SwiGLU expert on pre-converted bf16 weights.
// One block = 16 gathered tokens of expert e; 8 waves. Wave w computes one
// 16x16 SwiGLU tile per ff-chunk (stage A) and owns out columns
// [w*128, w*128+128) in 8 register-resident 16x16 f32 accumulators (stage B).
// W1/W3/x staged into LDS with global_load_async_to_lds_b128 (ASYNCcnt).
// -------------------------------------------------------------------------
__global__ __launch_bounds__(256) void moe_expert_pre(
    const uint16_t* __restrict__ xb,
    const uint16_t* __restrict__ wb1, const uint16_t* __restrict__ wb2,
    const uint16_t* __restrict__ wb3,
    const int* __restrict__ cnt, const int* __restrict__ tok,
    const float* __restrict__ wt,
    float* __restrict__ out)
{
    __shared__ alignas(16) __bf16 xs[16 * XSTRIDE];
    __shared__ alignas(16) __bf16 w1s[FF_CHUNK * WSTRIDE];
    __shared__ alignas(16) __bf16 w3s[FF_CHUNK * WSTRIDE];
    __shared__ alignas(16) __bf16 hs[16 * HSTRIDE];
    __shared__ int   tks[16];
    __shared__ float wts[16];

    const int e    = blockIdx.x >> 7;
    const int tile = blockIdx.x & 127;
    const int nt   = cnt[e];
    if (tile * 16 >= nt) return;            // uniform -> EXEC stays all-ones

    const int tid  = threadIdx.x;
    const int wave = tid >> 5;
    const int lane = tid & 31;
    const int nlo  = lane & 15;
    const int khi  = lane >> 4;

    if (tid < 16) {
        int idx = tile * 16 + tid;
        if (idx < nt) { tks[tid] = tok[e * CAP + idx]; wts[tid] = wt[e * CAP + idx]; }
        else          { tks[tid] = 0;                  wts[tid] = 0.f; }
    }
    __syncthreads();

    // ---- stage x rows (bf16 global -> LDS) via async-to-LDS, 128B/thread ----
    {
        int r  = tid >> 4;
        int c0 = (tid & 15) * 64;                     // bf16 elems
        const uint16_t* src = xb + (size_t)tks[r] * D_MODEL + c0;
        uint32_t dst = lds_off(&xs[r * XSTRIDE + c0]);
#pragma unroll
        for (int q = 0; q < 8; ++q)
            async_ld_b128(dst + 16 * q, src + 8 * q);
    }
    wait_async0();
    __syncthreads();

    const uint16_t* w1e = wb1 + (size_t)e * D_FF * D_MODEL;
    const uint16_t* w3e = wb3 + (size_t)e * D_FF * D_MODEL;
    const uint16_t* w2e = wb2 + (size_t)e * D_MODEL * D_FF;

    v8f accd[8];
#pragma unroll
    for (int i = 0; i < 8; ++i)
#pragma unroll
        for (int j = 0; j < 8; ++j) accd[i][j] = 0.f;

    for (int ff0 = 0; ff0 < D_FF; ff0 += FF_CHUNK) {
        v8f accg, accu;
#pragma unroll
        for (int j = 0; j < 8; ++j) { accg[j] = 0.f; accu[j] = 0.f; }

        // ---- stage A: gate/up GEMMs over K = D_MODEL ----
        for (int k0 = 0; k0 < D_MODEL; k0 += KSTEP) {
            __syncthreads();                 // previous slab reads complete
            {   // async stage: [FF_CHUNK x KSTEP] bf16 slab of W1 and W3
                int r  = tid >> 1;
                int cb = (tid & 1) * 16;     // bf16 elems (32B per thread/matrix)
                const uint16_t* s1 = w1e + (size_t)(ff0 + r) * D_MODEL + k0 + cb;
                const uint16_t* s3 = w3e + (size_t)(ff0 + r) * D_MODEL + k0 + cb;
                uint32_t d1 = lds_off(&w1s[r * WSTRIDE + cb]);
                uint32_t d3 = lds_off(&w3s[r * WSTRIDE + cb]);
                async_ld_b128(d1,      s1);
                async_ld_b128(d1 + 16, s1 + 8);
                async_ld_b128(d3,      s3);
                async_ld_b128(d3 + 16, s3 + 8);
            }
            wait_async0();
            __syncthreads();

            V16BF a;                          // A operand (x tile)
            {
                int kb = k0 + khi * 16;
                a.q[0] = *(const uint4*)&xs[nlo * XSTRIDE + kb];
                a.q[1] = *(const uint4*)&xs[nlo * XSTRIDE + kb + 8];
            }
            V16BF b1, b3;                     // B operands (W1^T / W3^T)
            {
                int row = wave * 16 + nlo;
                int kb  = khi * 16;
                const uint4* q1 = (const uint4*)&w1s[row * WSTRIDE + kb];
                const uint4* q3 = (const uint4*)&w3s[row * WSTRIDE + kb];
                b1.q[0] = q1[0]; b1.q[1] = q1[1];
                b3.q[0] = q3[0]; b3.q[1] = q3[1];
            }
            accg = __builtin_amdgcn_wmma_f32_16x16x32_bf16(
                false, a.v, false, b1.v, (short)0, accg, false, false);
            accu = __builtin_amdgcn_wmma_f32_16x16x32_bf16(
                false, a.v, false, b3.v, (short)0, accu, false, false);
        }

        // ---- SwiGLU elementwise, park h tile in LDS as bf16 ----
        __syncthreads();
#pragma unroll
        for (int i = 0; i < 8; ++i) {
            float g = accg[i], u = accu[i];
            float h = (g / (1.f + __expf(-g))) * u;
            int m = i + khi * 8;
            *(uint16_t*)&hs[m * HSTRIDE + wave * 16 + nlo] = f2bf(h);
        }
        __syncthreads();

        // ---- stage B: out += h_chunk @ W2[:, chunk]^T (bf16 direct loads) ----
#pragma unroll
        for (int ks = 0; ks < FF_CHUNK; ks += KSTEP) {
            V16BF ah;
            {
                int kb = ks + khi * 16;
                ah.q[0] = *(const uint4*)&hs[nlo * HSTRIDE + kb];
                ah.q[1] = *(const uint4*)&hs[nlo * HSTRIDE + kb + 8];
            }
#pragma unroll
            for (int dt = 0; dt < 8; ++dt) {
                int drow = wave * 128 + dt * 16 + nlo;
                const uint4* pw =
                    (const uint4*)(w2e + (size_t)drow * D_FF + ff0 + ks + khi * 16);
                V16BF bw;
                bw.q[0] = pw[0];
                bw.q[1] = pw[1];
                accd[dt] = __builtin_amdgcn_wmma_f32_16x16x32_bf16(
                    false, ah.v, false, bw.v, (short)0, accd[dt], false, false);
            }
        }
    }

    // ---- weighted scatter into out (<=2 contributions per element) ----
#pragma unroll
    for (int dt = 0; dt < 8; ++dt) {
        int d = wave * 128 + dt * 16 + nlo;
#pragma unroll
        for (int i = 0; i < 8; ++i) {
            int m = i + khi * 8;
            atomicAdd(&out[(size_t)tks[m] * D_MODEL + d], wts[m] * accd[dt][i]);
        }
    }
}

// -------------------------------------------------------------------------
// Kernel 2 (fallback): same algorithm, fp32 weights with in-kernel bf16
// conversion (used only if the workspace can't hold the bf16 weight copy).
// -------------------------------------------------------------------------
__global__ __launch_bounds__(256) void moe_expert_cvt(
    const float* __restrict__ x,
    const float* __restrict__ w1, const float* __restrict__ w2,
    const float* __restrict__ w3,
    const int* __restrict__ cnt, const int* __restrict__ tok,
    const float* __restrict__ wt,
    float* __restrict__ out)
{
    __shared__ alignas(16) __bf16 xs[16 * XSTRIDE];
    __shared__ alignas(16) __bf16 w1s[FF_CHUNK * WSTRIDE];
    __shared__ alignas(16) __bf16 w3s[FF_CHUNK * WSTRIDE];
    __shared__ alignas(16) __bf16 hs[16 * HSTRIDE];
    __shared__ int   tks[16];
    __shared__ float wts[16];

    const int e    = blockIdx.x >> 7;
    const int tile = blockIdx.x & 127;
    const int nt   = cnt[e];
    if (tile * 16 >= nt) return;

    const int tid  = threadIdx.x;
    const int wave = tid >> 5;
    const int lane = tid & 31;
    const int nlo  = lane & 15;
    const int khi  = lane >> 4;

    if (tid < 16) {
        int idx = tile * 16 + tid;
        if (idx < nt) { tks[tid] = tok[e * CAP + idx]; wts[tid] = wt[e * CAP + idx]; }
        else          { tks[tid] = 0;                  wts[tid] = 0.f; }
    }
    __syncthreads();

    {
        int r  = tid >> 4;
        int c0 = (tid & 15) * 64;
        const float4* src = (const float4*)(x + (size_t)tks[r] * D_MODEL + c0);
#pragma unroll
        for (int q = 0; q < 16; ++q) {
            float4 f = src[q];
            *(uint32_t*)&xs[r * XSTRIDE + c0 + 4 * q]     = pk_bf16(f.x, f.y);
            *(uint32_t*)&xs[r * XSTRIDE + c0 + 4 * q + 2] = pk_bf16(f.z, f.w);
        }
    }

    const float* w1e = w1 + (size_t)e * D_FF * D_MODEL;
    const float* w3e = w3 + (size_t)e * D_FF * D_MODEL;
    const float* w2e = w2 + (size_t)e * D_MODEL * D_FF;

    v8f accd[8];
#pragma unroll
    for (int i = 0; i < 8; ++i)
#pragma unroll
        for (int j = 0; j < 8; ++j) accd[i][j] = 0.f;

    for (int ff0 = 0; ff0 < D_FF; ff0 += FF_CHUNK) {
        v8f accg, accu;
#pragma unroll
        for (int j = 0; j < 8; ++j) { accg[j] = 0.f; accu[j] = 0.f; }

        for (int k0 = 0; k0 < D_MODEL; k0 += KSTEP) {
            __syncthreads();
            {
                int r  = tid >> 1;
                int cb = (tid & 1) * 16;
                const float4* s1 = (const float4*)(w1e + (size_t)(ff0 + r) * D_MODEL + k0 + cb);
                const float4* s3 = (const float4*)(w3e + (size_t)(ff0 + r) * D_MODEL + k0 + cb);
#pragma unroll
                for (int q = 0; q < 4; ++q) {
                    float4 f1 = s1[q], f3 = s3[q];
                    *(uint32_t*)&w1s[r * WSTRIDE + cb + 4 * q]     = pk_bf16(f1.x, f1.y);
                    *(uint32_t*)&w1s[r * WSTRIDE + cb + 4 * q + 2] = pk_bf16(f1.z, f1.w);
                    *(uint32_t*)&w3s[r * WSTRIDE + cb + 4 * q]     = pk_bf16(f3.x, f3.y);
                    *(uint32_t*)&w3s[r * WSTRIDE + cb + 4 * q + 2] = pk_bf16(f3.z, f3.w);
                }
            }
            __syncthreads();

            V16BF a;
            {
                int kb = k0 + khi * 16;
                a.q[0] = *(const uint4*)&xs[nlo * XSTRIDE + kb];
                a.q[1] = *(const uint4*)&xs[nlo * XSTRIDE + kb + 8];
            }
            V16BF b1, b3;
            {
                int row = wave * 16 + nlo;
                int kb  = khi * 16;
                const uint4* q1 = (const uint4*)&w1s[row * WSTRIDE + kb];
                const uint4* q3 = (const uint4*)&w3s[row * WSTRIDE + kb];
                b1.q[0] = q1[0]; b1.q[1] = q1[1];
                b3.q[0] = q3[0]; b3.q[1] = q3[1];
            }
            accg = __builtin_amdgcn_wmma_f32_16x16x32_bf16(
                false, a.v, false, b1.v, (short)0, accg, false, false);
            accu = __builtin_amdgcn_wmma_f32_16x16x32_bf16(
                false, a.v, false, b3.v, (short)0, accu, false, false);
        }

        __syncthreads();
#pragma unroll
        for (int i = 0; i < 8; ++i) {
            float g = accg[i], u = accu[i];
            float h = (g / (1.f + __expf(-g))) * u;
            int m = i + khi * 8;
            *(uint16_t*)&hs[m * HSTRIDE + wave * 16 + nlo] = f2bf(h);
        }
        __syncthreads();

#pragma unroll
        for (int ks = 0; ks < FF_CHUNK; ks += KSTEP) {
            V16BF ah;
            {
                int kb = ks + khi * 16;
                ah.q[0] = *(const uint4*)&hs[nlo * HSTRIDE + kb];
                ah.q[1] = *(const uint4*)&hs[nlo * HSTRIDE + kb + 8];
            }
#pragma unroll
            for (int dt = 0; dt < 8; ++dt) {
                int drow = wave * 128 + dt * 16 + nlo;
                const float4* pw =
                    (const float4*)(w2e + (size_t)drow * D_FF + ff0 + ks + khi * 16);
                V16BF bw;
#pragma unroll
                for (int q = 0; q < 4; ++q) {
                    float4 f = pw[q];
                    bw.u[2 * q]     = pk_bf16(f.x, f.y);
                    bw.u[2 * q + 1] = pk_bf16(f.z, f.w);
                }
                accd[dt] = __builtin_amdgcn_wmma_f32_16x16x32_bf16(
                    false, ah.v, false, bw.v, (short)0, accd[dt], false, false);
            }
        }
    }

#pragma unroll
    for (int dt = 0; dt < 8; ++dt) {
        int d = wave * 128 + dt * 16 + nlo;
#pragma unroll
        for (int i = 0; i < 8; ++i) {
            int m = i + khi * 8;
            atomicAdd(&out[(size_t)tks[m] * D_MODEL + d], wts[m] * accd[dt][i]);
        }
    }
}

// -------------------------------------------------------------------------
extern "C" void kernel_launch(void* const* d_in, const int* in_sizes, int n_in,
                              void* d_out, int out_size, void* d_ws, size_t ws_size,
                              hipStream_t stream) {
    (void)in_sizes; (void)n_in; (void)out_size;
    const float* x  = (const float*)d_in[0];
    const float* wr = (const float*)d_in[1];
    const float* w1 = (const float*)d_in[2];
    const float* w2 = (const float*)d_in[3];
    const float* w3 = (const float*)d_in[4];
    float* out = (float*)d_out;

    const size_t W_ELEMS = (size_t)N_EXP * D_FF * D_MODEL;   // 33.5M per matrix
    const size_t X_ELEMS = (size_t)T_TOK * D_MODEL;          // 2.1M

    char*  ws  = (char*)d_ws;
    size_t o_tok = 256;
    size_t o_wt  = o_tok + (size_t)N_EXP * CAP * sizeof(int);
    size_t o_xb  = (o_wt + (size_t)N_EXP * CAP * sizeof(float) + 255) & ~(size_t)255;
    size_t o_w1b = o_xb  + X_ELEMS * 2;
    size_t o_w2b = o_w1b + W_ELEMS * 2;
    size_t o_w3b = o_w2b + W_ELEMS * 2;
    size_t need  = o_w3b + W_ELEMS * 2;

    int*      cnt = (int*)ws;
    int*      tok = (int*)(ws + o_tok);
    float*    wt  = (float*)(ws + o_wt);
    uint16_t* xb  = (uint16_t*)(ws + o_xb);
    uint16_t* wb1 = (uint16_t*)(ws + o_w1b);
    uint16_t* wb2 = (uint16_t*)(ws + o_w2b);
    uint16_t* wb3 = (uint16_t*)(ws + o_w3b);

    hipMemsetAsync(cnt, 0, N_EXP * sizeof(int), stream);
    hipMemsetAsync(out, 0, (size_t)T_TOK * D_MODEL * sizeof(float), stream);

    moe_router<<<T_TOK / 8, 256, 0, stream>>>(x, wr, cnt, tok, wt);

    if (ws_size >= need) {
        // one-time fp32->bf16 pass; GEMMs then stream pure bf16 (L2-resident
        // per expert: ~25MB bf16 << 192MB L2)
        cvt_f32_bf16<<<2048, 256, 0, stream>>>((const float4*)x,  (uint2*)xb,
                                               (int)(X_ELEMS / 4));
        cvt_f32_bf16<<<2048, 256, 0, stream>>>((const float4*)w1, (uint2*)wb1,
                                               (int)(W_ELEMS / 4));
        cvt_f32_bf16<<<2048, 256, 0, stream>>>((const float4*)w2, (uint2*)wb2,
                                               (int)(W_ELEMS / 4));
        cvt_f32_bf16<<<2048, 256, 0, stream>>>((const float4*)w3, (uint2*)wb3,
                                               (int)(W_ELEMS / 4));
        moe_expert_pre<<<N_EXP * (T_TOK / 16), 256, 0, stream>>>(
            xb, wb1, wb2, wb3, cnt, tok, wt, out);
    } else {
        moe_expert_cvt<<<N_EXP * (T_TOK / 16), 256, 0, stream>>>(
            x, w1, w2, w3, cnt, tok, wt, out);
    }
}
